// MyEmoSubword_66451734003912
// MI455X (gfx1250) — matrix operand order repaired and checked by reading, not compiled, not run on hardware
//
#include <hip/hip_runtime.h>
#include <hip/hip_bf16.h>
#include <math.h>

// Problem dims (compile-time; all GEMM dims are multiples of the 128x64x32 tile)
#define B_   32
#define S_   1024
#define H_   768
#define L_   28
#define NH_  8
#define HD_  96
#define H2_  1536
#define BL_  (B_ * L_)   // 896

typedef __attribute__((ext_vector_type(16))) __bf16 v16bf;
typedef __attribute__((ext_vector_type(8)))  __bf16 v8bf;
typedef __attribute__((ext_vector_type(8)))  float  v8f;

// ---------------------------------------------------------------- reductions
__device__ __forceinline__ float wave_sum(float v) {
#pragma unroll
  for (int o = 16; o > 0; o >>= 1) v += __shfl_down(v, o, 32);
  return v;
}
__device__ __forceinline__ float wave_max(float v) {
#pragma unroll
  for (int o = 16; o > 0; o >>= 1) v = fmaxf(v, __shfl_down(v, o, 32));
  return v;
}
__device__ __forceinline__ float block_sum(float v, float* red, int nw) {
  const int lane = threadIdx.x & 31, w = threadIdx.x >> 5;
  v = wave_sum(v);
  if (lane == 0) red[w] = v;
  __syncthreads();
  if (w == 0) {
    float x = (lane < nw) ? red[lane] : 0.f;
    x = wave_sum(x);
    if (lane == 0) red[0] = x;
  }
  __syncthreads();
  float r = red[0];
  __syncthreads();
  return r;
}
__device__ __forceinline__ float block_max(float v, float* red, int nw) {
  const int lane = threadIdx.x & 31, w = threadIdx.x >> 5;
  v = wave_max(v);
  if (lane == 0) red[w] = v;
  __syncthreads();
  if (w == 0) {
    float x = (lane < nw) ? red[lane] : -3.0e38f;
    x = wave_max(x);
    if (lane == 0) red[0] = x;
  }
  __syncthreads();
  float r = red[0];
  __syncthreads();
  return r;
}

// ---------------------------------------------------------------- fp32 -> bf16
__global__ __launch_bounds__(256) void cvt_bf16_kernel(
    const float* __restrict__ src, __bf16* __restrict__ dst, size_t n) {
  size_t i = (size_t)blockIdx.x * blockDim.x + threadIdx.x;
  const size_t stride = (size_t)gridDim.x * blockDim.x;
  for (; i < n; i += stride) dst[i] = (__bf16)src[i];
}

// ---------------------------------------------------------------- span pooling
__global__ __launch_bounds__(256) void pool_kernel(
    const float* __restrict__ hidden, const int* __restrict__ lpos,
    float* __restrict__ labF, __bf16* __restrict__ labB) {
  const int bl = blockIdx.x;                 // b*L + l
  const int b = bl / L_;
  const int start = lpos[bl * 2];
  const int len   = lpos[bl * 2 + 1];
  const float inv = 1.0f / (float)max(len, 1);
  for (int h = threadIdx.x; h < H_; h += 256) {
    float s = 0.f;
    for (int p2 = 0; p2 < len; ++p2)
      s += hidden[((size_t)b * S_ + start + p2) * H_ + h];
    const float v = s * inv;
    labF[(size_t)bl * H_ + h] = v;
    labB[(size_t)bl * H_ + h] = (__bf16)v;
  }
}

// ---------------------------------------------------------------- WMMA GEMM
// C[M,N] = A[M,K] @ W[K,N] + bias, row-major; A,W bf16; fp32 accum via
// v_wmma_f32_16x16x32_bf16. Block: 128 threads = 4 waves; tile 128x64, K-step 32.
// Wave w owns rows [32w, 32w+32) as two 16-row fragments; each B fragment is
// reused by both -> 8 wmma per wave per K-step.
//
// Software-pipelined double buffering: tile k+1 is fetched into REGISTERS while
// the 8 WMMAs of tile k run from LDS buffer (k&1); registers are then spilled
// to buffer ((k+1)&1) behind the compute. One barrier per K-step.
//
// LDS layouts make every fragment load a wide ds_load_b128:
//  - sA  row-major [m][k], stride 40 halves (80B = 20 banks; 20*hl mod 64 is
//    injective for hl<16 -> conflict-free within a half-wave). A-fragment per
//    ISA 7.12.2 = two contiguous 8-half runs: K = hi*8+(0..7), 16+hi*8+(0..7).
//  - sWt TRANSPOSED [n][k], stride 40 halves. B-fragment lane layout
//    (lane&15 = N, K = hi*8+(0..7), 16+hi*8+(0..7), same as A) is then two contiguous 8-half runs.
// EPI: 0 = f32 out (+bias), 1 = bf16 out (+bias), 2 = exact-GELU -> bf16 out.
#define LDS_STRIDE 40
template <int EPI>
__global__ __launch_bounds__(128) void gemm_bf16_wmma(
    const __bf16* __restrict__ A, const __bf16* __restrict__ W,
    const float* __restrict__ bias, float* __restrict__ outF,
    __bf16* __restrict__ outB, int M, int N, int K) {
  __shared__ __align__(16) __bf16 sA[2][128 * LDS_STRIDE];
  __shared__ __align__(16) __bf16 sWt[2][64 * LDS_STRIDE];
  const int tid  = threadIdx.x;
  const int wave = tid >> 5;
  const int lane = tid & 31;
  const int hl   = lane & 15;
  const int hi   = lane >> 4;
  const int m0   = blockIdx.y * 128;
  const int n0   = blockIdx.x * 64;
  const int nk   = K >> 5;  // K / 32

  v8f acc[2][4];
#pragma unroll
  for (int mi = 0; mi < 2; ++mi)
#pragma unroll
    for (int i = 0; i < 4; ++i)
#pragma unroll
      for (int j = 0; j < 8; ++j) acc[mi][i][j] = 0.f;

  // Register staging buffers (exec-uniform: loop bounds are compile-time).
  v8bf ra[4], rw[2];
  auto load_tile = [&](int k0) {
#pragma unroll
    for (int j = 0; j < 4; ++j) {           // A tile: 128x32, 16B chunk each
      const int id = tid + j * 128, r = id >> 2, q = id & 3;
      ra[j] = *(const v8bf*)(A + (size_t)(m0 + r) * K + k0 + q * 8);
    }
#pragma unroll
    for (int j = 0; j < 2; ++j) {           // W tile: 32x64, 16B chunk each
      const int id = tid + j * 128, r = id >> 3, q = id & 7;
      rw[j] = *(const v8bf*)(W + (size_t)(k0 + r) * N + n0 + q * 8);
    }
  };
  auto store_tile = [&](int buf) {
#pragma unroll
    for (int j = 0; j < 4; ++j) {
      const int id = tid + j * 128, r = id >> 2, q = id & 3;
      *(v8bf*)&sA[buf][r * LDS_STRIDE + q * 8] = ra[j];
    }
#pragma unroll
    for (int j = 0; j < 2; ++j) {           // transposed scatter (amortized)
      const int id = tid + j * 128, r = id >> 3, q = id & 7;
#pragma unroll
      for (int e = 0; e < 8; ++e) sWt[buf][(q * 8 + e) * LDS_STRIDE + r] = rw[j][e];
    }
  };

  load_tile(0);
  store_tile(0);

  for (int kt = 0; kt < nk; ++kt) {
    __syncthreads();
    const int buf = kt & 1;
    const bool more = (kt + 1) < nk;
    if (more) load_tile((kt + 1) << 5);     // global loads issue early

    v16bf a[2];
#pragma unroll
    for (int mi = 0; mi < 2; ++mi) {
      const int am = wave * 32 + mi * 16 + hl;
      const v8bf alo = *(const v8bf*)&sA[buf][am * LDS_STRIDE + hi * 8];
      const v8bf ahi = *(const v8bf*)&sA[buf][am * LDS_STRIDE + 16 + hi * 8];
      a[mi] = __builtin_shufflevector(alo, ahi, 0, 1, 2, 3, 4, 5, 6, 7,
                                      8, 9, 10, 11, 12, 13, 14, 15);
    }
#pragma unroll
    for (int nc = 0; nc < 4; ++nc) {
      const int bn = (nc * 16 + hl) * LDS_STRIDE + hi * 8;
      const v8bf blo = *(const v8bf*)&sWt[buf][bn];
      const v8bf bhi = *(const v8bf*)&sWt[buf][bn + 16];
      const v16bf bf = __builtin_shufflevector(blo, bhi, 0, 1, 2, 3, 4, 5, 6, 7,
                                               8, 9, 10, 11, 12, 13, 14, 15);
      acc[0][nc] = __builtin_amdgcn_wmma_f32_16x16x32_bf16(
          false, a[0], false, bf, (short)0, acc[0][nc], false, false);
      acc[1][nc] = __builtin_amdgcn_wmma_f32_16x16x32_bf16(
          false, a[1], false, bf, (short)0, acc[1][nc], false, false);
    }
    if (more) store_tile((kt + 1) & 1);     // spill next tile behind compute
  }

#pragma unroll
  for (int mi = 0; mi < 2; ++mi) {
#pragma unroll
    for (int nc = 0; nc < 4; ++nc) {
#pragma unroll
      for (int r = 0; r < 8; ++r) {
        const int row = m0 + wave * 32 + mi * 16 + hi * 8 + r;
        const int col = n0 + nc * 16 + hl;
        float vv = acc[mi][nc][r] + bias[col];
        if (EPI == 2) vv = 0.5f * vv * (1.f + erff(vv * 0.70710678118f));
        if (EPI == 0)
          outF[(size_t)row * N + col] = vv;
        else
          outB[(size_t)row * N + col] = (__bf16)vv;
      }
    }
  }
}

// ---------------------------------------------------------------- attention
// One block per (b, head). L=28 query rows; softmax over S=1024 keys.
// K/V read as bf16 (L2-resident), math in fp32; K rows read as 16B vectors.
__global__ __launch_bounds__(256) void attn_kernel(
    const float* __restrict__ q, const __bf16* __restrict__ k,
    const __bf16* __restrict__ v, float* __restrict__ ctx) {
  __shared__ float sq[HD_];
  __shared__ float sp[S_];
  __shared__ float red[8];
  const int b  = blockIdx.x / NH_;
  const int nh = blockIdx.x % NH_;
  const int tid = threadIdx.x;
  const float scale = 0.1020620726f;  // 1/sqrt(96)

  for (int l = 0; l < L_; ++l) {
    for (int d = tid; d < HD_; d += 256)
      sq[d] = q[((size_t)(b * L_ + l)) * H_ + nh * HD_ + d];
    __syncthreads();

    float sc[4];
    float lmax = -3.0e38f;
#pragma unroll
    for (int t = 0; t < 4; ++t) {
      const int s = tid + t * 256;
      const __bf16* kp = k + ((size_t)b * S_ + s) * H_ + nh * HD_;
      float acc = 0.f;
#pragma unroll
      for (int dv = 0; dv < HD_ / 8; ++dv) {   // 12 x 16B vector loads
        const v8bf kv = *(const v8bf*)(kp + dv * 8);
#pragma unroll
        for (int e = 0; e < 8; ++e) acc += sq[dv * 8 + e] * (float)kv[e];
      }
      sc[t] = acc * scale;
      lmax = fmaxf(lmax, sc[t]);
    }
    const float bmax = block_max(lmax, red, 8);
    float ls = 0.f;
#pragma unroll
    for (int t = 0; t < 4; ++t) {
      const float p = __expf(sc[t] - bmax);
      sp[tid + t * 256] = p;
      ls += p;
    }
    const float bsum = block_sum(ls, red, 8);  // barrier also publishes sp[]
    const float inv = 1.f / bsum;

    if (tid < HD_) {
      const __bf16* vp = v + (size_t)b * S_ * H_ + nh * HD_ + tid;
      float acc = 0.f;
      for (int s = 0; s < S_; ++s) acc += sp[s] * (float)vp[(size_t)s * H_];
      ctx[((size_t)(b * L_ + l)) * H_ + nh * HD_ + tid] = acc * inv;
    }
    __syncthreads();
  }
}

// ---------------------------------------------------------------- residual+LN
template <bool WB>
__global__ __launch_bounds__(256) void add_ln_kernel(
    const float* __restrict__ a, const float* __restrict__ b,
    const float* __restrict__ g, const float* __restrict__ beta,
    float* __restrict__ outF, __bf16* __restrict__ outB) {
  __shared__ float red[8];
  const int row = blockIdx.x;
  const size_t base = (size_t)row * H_;
  float x[3];
  float s = 0.f;
#pragma unroll
  for (int i = 0; i < 3; ++i) {
    const int h = threadIdx.x + i * 256;
    x[i] = a[base + h] + b[base + h];
    s += x[i];
  }
  const float mu = block_sum(s, red, 8) * (1.f / H_);
  float vv = 0.f;
#pragma unroll
  for (int i = 0; i < 3; ++i) {
    const float d = x[i] - mu;
    vv += d * d;
  }
  const float var = block_sum(vv, red, 8) * (1.f / H_);
  const float rstd = rsqrtf(var + 1e-5f);
#pragma unroll
  for (int i = 0; i < 3; ++i) {
    const int h = threadIdx.x + i * 256;
    const float y = (x[i] - mu) * rstd * g[h] + beta[h];
    outF[base + h] = y;
    if (WB) outB[base + h] = (__bf16)y;
  }
}

// ---------------------------------------------------------------- logits
__global__ __launch_bounds__(128) void logits_kernel(
    const float* __restrict__ x, const float* __restrict__ wc,
    const float* __restrict__ bc, float* __restrict__ out_logits) {
  __shared__ float red[4];
  const int row = blockIdx.x;
  float s = 0.f;
  for (int h = threadIdx.x; h < H_; h += 128) s += x[(size_t)row * H_ + h] * wc[h];
  s = block_sum(s, red, 4);
  if (threadIdx.x == 0) out_logits[row] = s + bc[0];
}

// ---------------------------------------------------------------- loss
// (1-alpha)*mean BCE-with-logits + alpha * mean over (pos,neg) pairs of
// relu(l_neg - l_pos + 0.3). 896 logits -> 896^2 pairs, one block.
__global__ __launch_bounds__(256) void loss_kernel(
    const float* __restrict__ logits, const float* __restrict__ labels,
    float* __restrict__ out_loss) {
  __shared__ float sl[BL_], sy[BL_];
  __shared__ float red[8];
  const int tid = threadIdx.x;
  for (int i = tid; i < BL_; i += 256) {
    sl[i] = logits[i];
    sy[i] = labels[i];
  }
  __syncthreads();
  float bce = 0.f, np = 0.f, nn = 0.f, cs = 0.f;
  for (int i = tid; i < BL_; i += 256) {
    const float l = sl[i], y = sy[i];
    bce += fmaxf(l, 0.f) - l * y + log1pf(__expf(-fabsf(l)));
    if (y > 0.5f) np += 1.f; else nn += 1.f;
  }
  for (int i = tid; i < BL_; i += 256) {
    if (sy[i] > 0.5f) {
      const float li = sl[i];
      for (int j = 0; j < BL_; ++j)
        if (sy[j] < 0.5f) cs += fmaxf(sl[j] - li + 0.3f, 0.f);
    }
  }
  bce = block_sum(bce, red, 8);
  np  = block_sum(np,  red, 8);
  nn  = block_sum(nn,  red, 8);
  cs  = block_sum(cs,  red, 8);
  if (tid == 0) {
    const float denom = np * nn;
    const float corr = (denom > 0.f) ? cs / fmaxf(denom, 1.f) : 0.f;
    out_loss[0] = 0.8f * (bce / (float)BL_) + 0.2f * corr;
  }
}

// ---------------------------------------------------------------- launch
extern "C" void kernel_launch(void* const* d_in, const int* in_sizes, int n_in,
                              void* d_out, int out_size, void* d_ws, size_t ws_size,
                              hipStream_t stream) {
  (void)in_sizes; (void)n_in; (void)out_size; (void)ws_size;
  const float* hidden = (const float*)d_in[0];
  const float* labels = (const float*)d_in[1];
  const int*   lpos   = (const int*)d_in[2];
  const float* Wq = (const float*)d_in[3];  const float* bq  = (const float*)d_in[4];
  const float* Wk = (const float*)d_in[5];  const float* bk  = (const float*)d_in[6];
  const float* Wv = (const float*)d_in[7];  const float* bv  = (const float*)d_in[8];
  const float* Wo = (const float*)d_in[9];  const float* bo  = (const float*)d_in[10];
  const float* g1 = (const float*)d_in[11]; const float* be1 = (const float*)d_in[12];
  const float* W1 = (const float*)d_in[13]; const float* bf1 = (const float*)d_in[14];
  const float* W2 = (const float*)d_in[15]; const float* bf2 = (const float*)d_in[16];
  const float* g2 = (const float*)d_in[17]; const float* be2 = (const float*)d_in[18];
  const float* wc = (const float*)d_in[19]; const float* bc  = (const float*)d_in[20];
  float* out = (float*)d_out;  // out[0]=loss, out[1..896]=logits

  // ---- workspace carve (bump allocator, 256B aligned)
  char* p = (char*)d_ws;
  auto alloc = [&](size_t bytes) -> char* {
    char* r = p;
    p += (bytes + 255) & ~(size_t)255;
    return r;
  };
  __bf16* hidB = (__bf16*)alloc((size_t)B_ * S_ * H_ * 2);   // 48 MB
  __bf16* WqB  = (__bf16*)alloc((size_t)H_ * H_ * 2);
  __bf16* WkB  = (__bf16*)alloc((size_t)H_ * H_ * 2);
  __bf16* WvB  = (__bf16*)alloc((size_t)H_ * H_ * 2);
  __bf16* WoB  = (__bf16*)alloc((size_t)H_ * H_ * 2);
  __bf16* W1B  = (__bf16*)alloc((size_t)H_ * H2_ * 2);
  __bf16* W2B  = (__bf16*)alloc((size_t)H2_ * H_ * 2);
  float*  labF = (float*)alloc((size_t)BL_ * H_ * 4);
  __bf16* labB = (__bf16*)alloc((size_t)BL_ * H_ * 2);
  float*  qF   = (float*)alloc((size_t)BL_ * H_ * 4);
  __bf16* kB   = (__bf16*)alloc((size_t)B_ * S_ * H_ * 2);   // 48 MB
  __bf16* vB   = (__bf16*)alloc((size_t)B_ * S_ * H_ * 2);   // 48 MB
  float*  ctxF = (float*)alloc((size_t)BL_ * H_ * 4);
  __bf16* ctxB = (__bf16*)alloc((size_t)BL_ * H_ * 2);
  float*  aoF  = (float*)alloc((size_t)BL_ * H_ * 4);
  float*  xF   = (float*)alloc((size_t)BL_ * H_ * 4);
  __bf16* xB   = (__bf16*)alloc((size_t)BL_ * H_ * 2);
  __bf16* h1B  = (__bf16*)alloc((size_t)BL_ * H2_ * 2);
  float*  zF   = (float*)alloc((size_t)BL_ * H_ * 4);
  float*  x2F  = (float*)alloc((size_t)BL_ * H_ * 4);

  // ---- 1. casts to bf16
  cvt_bf16_kernel<<<4096, 256, 0, stream>>>(hidden, hidB, (size_t)B_ * S_ * H_);
  cvt_bf16_kernel<<<512, 256, 0, stream>>>(Wq, WqB, (size_t)H_ * H_);
  cvt_bf16_kernel<<<512, 256, 0, stream>>>(Wk, WkB, (size_t)H_ * H_);
  cvt_bf16_kernel<<<512, 256, 0, stream>>>(Wv, WvB, (size_t)H_ * H_);
  cvt_bf16_kernel<<<512, 256, 0, stream>>>(Wo, WoB, (size_t)H_ * H_);
  cvt_bf16_kernel<<<1024, 256, 0, stream>>>(W1, W1B, (size_t)H_ * H2_);
  cvt_bf16_kernel<<<1024, 256, 0, stream>>>(W2, W2B, (size_t)H2_ * H_);

  // ---- 2. span mean-pooling
  pool_kernel<<<BL_, 256, 0, stream>>>(hidden, lpos, labF, labB);

  // ---- 3. projections (WMMA). K/V are the 77-GFLOP bulk.
  gemm_bf16_wmma<0><<<dim3(H_ / 64, BL_ / 128), 128, 0, stream>>>(
      labB, WqB, bq, qF, nullptr, BL_, H_, H_);
  gemm_bf16_wmma<1><<<dim3(H_ / 64, (B_ * S_) / 128), 128, 0, stream>>>(
      hidB, WkB, bk, nullptr, kB, B_ * S_, H_, H_);
  gemm_bf16_wmma<1><<<dim3(H_ / 64, (B_ * S_) / 128), 128, 0, stream>>>(
      hidB, WvB, bv, nullptr, vB, B_ * S_, H_, H_);

  // ---- 4. cross-attention (bf16 K/V are L2-resident: 96 MB < 192 MB L2)
  attn_kernel<<<B_ * NH_, 256, 0, stream>>>(qF, kB, vB, ctxF);

  // ---- 5. output projection + residual + LN
  cvt_bf16_kernel<<<2688, 256, 0, stream>>>(ctxF, ctxB, (size_t)BL_ * H_);
  gemm_bf16_wmma<0><<<dim3(H_ / 64, BL_ / 128), 128, 0, stream>>>(
      ctxB, WoB, bo, aoF, nullptr, BL_, H_, H_);
  add_ln_kernel<true><<<BL_, 256, 0, stream>>>(aoF, labF, g1, be1, xF, xB);

  // ---- 6. FFN with fused exact-GELU epilogue
  gemm_bf16_wmma<2><<<dim3(H2_ / 64, BL_ / 128), 128, 0, stream>>>(
      xB, W1B, bf1, nullptr, h1B, BL_, H2_, H_);
  gemm_bf16_wmma<0><<<dim3(H_ / 64, BL_ / 128), 128, 0, stream>>>(
      h1B, W2B, bf2, zF, nullptr, BL_, H_, H2_);
  add_ln_kernel<false><<<BL_, 256, 0, stream>>>(zF, xF, g2, be2, x2F, nullptr);

  // ---- 7. logits + joint loss
  logits_kernel<<<BL_, 128, 0, stream>>>(x2F, wc, bc, out + 1);
  loss_kernel<<<1, 256, 0, stream>>>(out + 1, labels, out);
}
